// GAT_10917806866470
// MI455X (gfx1250) — compile-verified, matrix operand order checked by reading, not run on hardware
//
#include <hip/hip_runtime.h>
#include <math.h>

#define LRELU 0.2f
#define SM_EPS 1e-16f
#define BN_EPS 1e-5f

typedef __attribute__((ext_vector_type(16))) __bf16 bf16x16;
typedef __attribute__((ext_vector_type(8)))  float  f32x8;

union Frag {
    unsigned short u[16];
    uint4 q[2];
    bf16x16 v;
};

__device__ __forceinline__ unsigned short f2bf(float f) {
    unsigned u = __float_as_uint(f);
    unsigned r = u + 0x7FFFu + ((u >> 16) & 1u);   // round-to-nearest-even
    return (unsigned short)(r >> 16);
}

__device__ __forceinline__ void atomicMaxF(float* addr, float v) {
    if (v >= 0.0f) atomicMax((int*)addr, __float_as_int(v));
    else           atomicMin((unsigned int*)addr, __float_as_uint(v));
}

typedef __attribute__((address_space(3))) const void lds_cv_t;
__device__ __forceinline__ unsigned lds_offset(const void* p) {
    return (unsigned)(unsigned long long)(lds_cv_t*)p;
}

// ---------------------------------------------------------------- fill
__global__ void fill_f32(float* p, float v, long n) {
    long i = (long)blockIdx.x * blockDim.x + threadIdx.x;
    long stride = (long)gridDim.x * blockDim.x;
    for (; i < n; i += stride) p[i] = v;
}

// ---------------------------------------------------------------- f32 -> bf16 (packed writes)
__global__ void cvt_bf16(const float* __restrict__ in, unsigned* __restrict__ out, long npairs) {
    long i = (long)blockIdx.x * blockDim.x + threadIdx.x;
    if (i >= npairs) return;
    float a = in[i * 2], b = in[i * 2 + 1];
    out[i] = (unsigned)f2bf(a) | ((unsigned)f2bf(b) << 16);
}

// --------------------------------------------------- exact percentile (radix select on |s| bits)
__global__ void radix_select(const float* s1, long n1, const float* s2, long n2,
                             long k, float* thr_out) {
    __shared__ unsigned cnt[256];
    __shared__ unsigned prefix;
    __shared__ long kk;
    int tid = threadIdx.x;
    if (tid == 0) { prefix = 0u; kk = k; }
    __syncthreads();
    long n = n1 + n2;
    for (int pass = 3; pass >= 0; --pass) {
        cnt[tid] = 0u;
        __syncthreads();
        unsigned prefmask = (pass == 3) ? 0u : (0xFFFFFFFFu << ((pass + 1) * 8));
        int shift = pass * 8;
        unsigned pfx = prefix;
        for (long i = tid; i < n; i += 256) {
            float v = (i < n1) ? s1[i] : s2[i - n1];
            unsigned b = __float_as_uint(fabsf(v));
            if ((b & prefmask) == pfx) atomicAdd(&cnt[(b >> shift) & 255u], 1u);
        }
        __syncthreads();
        if (tid == 0) {
            long c = 0;
            for (int j = 0; j < 256; ++j) {
                if (c + (long)cnt[j] > kk) { prefix |= ((unsigned)j << shift); kk -= c; break; }
                c += (long)cnt[j];
            }
        }
        __syncthreads();
    }
    if (tid == 0) *thr_out = __uint_as_float(prefix);
}

// ------------------------------------- mask + pack W into WMMA B-fragment layout (bf16)
// pack[((kc*CT + ct)*32 + lane)*16 + e] = Wm[kc*32 + (lane>=16?16:0) + e][ct*16 + (lane&15)]
__global__ void pack_W(const float* w, const float* s, const float* thr,
                       int F, int HD, int CT, unsigned short* pack) {
    int idx = blockIdx.x * blockDim.x + threadIdx.x;
    int total = (F / 32) * CT * 32;
    if (idx >= total) return;
    int lane = idx & 31;
    int t = idx >> 5;
    int ct = t % CT;
    int kc = t / CT;
    float th = *thr;
    int col = ct * 16 + (lane & 15);
    int kb = kc * 32 + ((lane & 16) ? 16 : 0);
    unsigned short* dst = pack + (long)idx * 16;
    for (int e = 0; e < 16; ++e) {
        int f = kb + e;
        float sv = s[(long)f * HD + col];
        float wv = (fabsf(sv) < th) ? 0.0f : w[(long)f * HD + col];
        dst[e] = f2bf(wv);
    }
}

// ---------------------------------------------------------------- h = x @ Wm via WMMA bf16
// 64 rows per block (4 row tiles), CT col-tile waves; x pre-converted to bf16.
#define XSTRIDE 136  // bf16 elements per LDS row (272B: 16B aligned, 4-bank shift/row)
#define ROWT 4

__global__ __launch_bounds__(320) void gemm_wmma(const unsigned short* __restrict__ xbf,
                                                 const unsigned short* __restrict__ packW,
                                                 float* __restrict__ h,
                                                 int N, int HD, int CT) {
    __shared__ __align__(16) unsigned short lx[16 * ROWT * XSTRIDE];
    int row0 = blockIdx.x * (16 * ROWT);
    int tid = threadIdx.y * 32 + threadIdx.x;
    int nth = 32 * CT;

    // async-copy 64x128 bf16 tile: 16 bytes per lane-slot, no VGPR round trip
    unsigned lbase = lds_offset(&lx[0]);
    for (int i = tid; i < 16 * ROWT * 16; i += nth) {
        int r = i >> 4, seg = i & 15;
        unsigned loff = lbase + (unsigned)(r * (XSTRIDE * 2) + seg * 16);
        const unsigned short* gp = xbf + ((long)(row0 + r) * 128 + seg * 8);
        asm volatile("global_load_async_to_lds_b128 %0, %1, off"
                     :: "v"(loff), "v"(gp) : "memory");
    }
    asm volatile("s_wait_asynccnt 0x0" ::: "memory");
    __syncthreads();

    int ct = threadIdx.y;
    int lane = threadIdx.x;
    int r = lane & 15;
    int khalf = (lane & 16) ? 16 : 0;

    f32x8 acc[ROWT];
    for (int t = 0; t < ROWT; ++t) acc[t] = {0.f, 0.f, 0.f, 0.f, 0.f, 0.f, 0.f, 0.f};

    const uint4* bp = (const uint4*)(packW + ((long)ct * 32 + lane) * 16);
    for (int kc = 0; kc < 4; ++kc) {
        Frag b;
        b.q[0] = bp[0]; b.q[1] = bp[1];
        bp += (long)CT * 32 * 2;           // next k-chunk (32 bytes per lane-slot)
        int kb = kc * 32 + khalf;
        for (int t = 0; t < ROWT; ++t) {
            Frag a;
            const uint4* ap = (const uint4*)&lx[(t * 16 + r) * XSTRIDE + kb];
            a.q[0] = ap[0]; a.q[1] = ap[1];
            acc[t] = __builtin_amdgcn_wmma_f32_16x16x32_bf16(false, a.v, false, b.v,
                                                             (short)0, acc[t], false, false);
        }
    }

    int col = ct * 16 + (lane & 15);
    int rsub = (lane & 16) ? 8 : 0;
    for (int t = 0; t < ROWT; ++t) {
        int rb = row0 + t * 16 + rsub;
        for (int i = 0; i < 8; ++i) {
            int rr = rb + i;
            if (rr < N) h[(long)rr * HD + col] = acc[t][i];
        }
    }
}

// ---------------------------------------------------------------- attention node scores
__global__ void attn_scores(const float* __restrict__ h,
                            const float* __restrict__ att_s, const float* __restrict__ att_d,
                            float* a_src, float* a_dst, int N, int H, int D) {
    int idx = blockIdx.x * blockDim.x + threadIdx.x;
    if (idx >= N * H) return;
    int n = idx / H, hh = idx - n * H;
    const float* hp = h + (long)n * H * D + hh * D;
    const float* as = att_s + hh * D;
    const float* ad = att_d + hh * D;
    float ss = 0.f, sd = 0.f;
    for (int d = 0; d < D; ++d) { float v = hp[d]; ss += v * as[d]; sd += v * ad[d]; }
    a_src[idx] = ss;
    a_dst[idx] = sd;
}

// ---------------------------------------------------------------- edge softmax phases
__global__ void edge_max(const int* __restrict__ src, const int* __restrict__ dst,
                         const float* __restrict__ a_s, const float* __restrict__ a_d,
                         float* m, long E, int H) {
    long e = (long)blockIdx.x * blockDim.x + threadIdx.x;
    if (e >= E) return;
    int s = src[e], d = dst[e];
    for (int hh = 0; hh < H; ++hh) {
        float v = a_s[(long)s * H + hh] + a_d[(long)d * H + hh];
        v = (v > 0.f) ? v : LRELU * v;
        atomicMaxF(&m[(long)d * H + hh], v);
    }
}

__global__ void m_fix(float* m, long n) {
    long i = (long)blockIdx.x * blockDim.x + threadIdx.x;
    if (i >= n) return;
    float v = m[i];
    if (!(v > -3.3e38f)) m[i] = 0.0f;
}

__global__ void edge_den(const int* __restrict__ src, const int* __restrict__ dst,
                         const float* __restrict__ a_s, const float* __restrict__ a_d,
                         const float* __restrict__ m, float* den, long E, int H) {
    long e = (long)blockIdx.x * blockDim.x + threadIdx.x;
    if (e >= E) return;
    int s = src[e], d = dst[e];
    for (int hh = 0; hh < H; ++hh) {
        float v = a_s[(long)s * H + hh] + a_d[(long)d * H + hh];
        v = (v > 0.f) ? v : LRELU * v;
        float ex = __expf(v - m[(long)d * H + hh]);
        atomicAdd(&den[(long)d * H + hh], ex);
    }
}

// one wave per edge; lane owns HD/32 contiguous dims (stays within one head)
__global__ __launch_bounds__(256) void edge_agg(const int* __restrict__ src,
                                                const int* __restrict__ dst,
                                                const float* __restrict__ a_s,
                                                const float* __restrict__ a_d,
                                                const float* __restrict__ m,
                                                const float* __restrict__ den,
                                                const float* __restrict__ h,
                                                float* out, long E, int H, int HD,
                                                int dper, float scale, int ostride, int omod) {
    long e = (long)blockIdx.x * 8 + (threadIdx.x >> 5);
    if (e >= E) return;
    int lane = threadIdx.x & 31;
    int s = src[e], d = dst[e];
    int hh = lane / (32 / H);
    float v = a_s[(long)s * H + hh] + a_d[(long)d * H + hh];
    v = (v > 0.f) ? v : LRELU * v;
    float alpha = __expf(v - m[(long)d * H + hh]) / (den[(long)d * H + hh] + SM_EPS);
    float w = alpha * scale;
    int c0 = lane * dper;
    const float* hp = h + (long)s * HD + c0;
    float* op = out + (long)d * ostride + (c0 % omod);
    for (int j = 0; j < dper; ++j) atomicAdd(&op[j], w * hp[j]);
}

// ---------------------------------------------------------------- batch norm (C == 128)
__global__ void bn_stats(const float* __restrict__ x, float* sums, float* sumsq, int N) {
    int col = threadIdx.x & 127;
    int r = blockIdx.x * 2 + (threadIdx.x >> 7);
    float s = 0.f, q = 0.f;
    for (; r < N; r += gridDim.x * 2) {
        float v = x[(long)r * 128 + col];
        s += v; q += v * v;
    }
    atomicAdd(&sums[col], s);
    atomicAdd(&sumsq[col], q);
}

__global__ void bn_apply(float* x, const float* sums, const float* sumsq,
                         const float* g, const float* b, int N) {
    long idx = (long)blockIdx.x * blockDim.x + threadIdx.x;
    if (idx >= (long)N * 128) return;
    int col = (int)(idx & 127);
    float mu = sums[col] / (float)N;
    float var = sumsq[col] / (float)N - mu * mu;
    float v = (x[idx] - mu) * rsqrtf(var + BN_EPS) * g[col] + b[col];
    x[idx] = (v > 0.f) ? v : 0.f;
}

// ----------------------------------------------------------------------------------------
extern "C" void kernel_launch(void* const* d_in, const int* in_sizes, int n_in,
                              void* d_out, int out_size, void* d_ws, size_t ws_size,
                              hipStream_t stream) {
    const float* d_x    = (const float*)d_in[0];
    const int*   d_ei   = (const int*)d_in[1];
    const float* d_w01  = (const float*)d_in[2];
    const float* d_s01  = (const float*)d_in[3];
    const float* d_att01= (const float*)d_in[4];
    const float* d_w2   = (const float*)d_in[5];
    const float* d_s2   = (const float*)d_in[6];
    const float* d_att2 = (const float*)d_in[7];
    const float* d_g    = (const float*)d_in[8];
    const float* d_b    = (const float*)d_in[9];

    const int  N    = in_sizes[0] / 128;
    const int  Npad = ((N + 63) / 64) * 64;
    const long E    = in_sizes[1] / 2;
    const long n1   = in_sizes[3];       // s01 elems
    const long n2   = in_sizes[6];       // s2 elems
    const int  H    = 4;

    // torch-style percentile index: k = round_half_even(0.5*(n-1))
    long n = n1 + n2;
    double pos = 0.5 * (double)(n - 1);
    double fl = floor(pos), fr = pos - fl;
    long k;
    if (fr > 0.5) k = (long)fl + 1;
    else if (fr < 0.5) k = (long)fl;
    else k = (((long)fl & 1) == 0) ? (long)fl : (long)fl + 1;

    // workspace layout
    char* ws = (char*)d_ws;
    size_t off = 0;
    auto take = [&](size_t bytes) { char* p = ws + off; off += (bytes + 255) & ~(size_t)255; return p; };
    float*          thr    = (float*)take(4);
    float*          bnsum  = (float*)take(128 * 4);
    float*          bnsq   = (float*)take(128 * 4);
    unsigned short* packw  = (unsigned short*)take(4 * 10 * 32 * 16 * 2); // max layer2
    float*          a_src  = (float*)take((size_t)N * H * 4);
    float*          a_dst  = (float*)take((size_t)N * H * 4);
    float*          mbuf   = (float*)take((size_t)N * H * 4);
    float*          den    = (float*)take((size_t)N * H * 4);
    unsigned short* xbf    = (unsigned short*)take((size_t)Npad * 128 * 2);
    float*          hbuf   = (float*)take((size_t)N * 160 * 4);
    float*          xA     = (float*)take((size_t)N * 128 * 4);
    float*          xB     = (float*)take((size_t)N * 128 * 4);
    (void)ws_size;

    const int* e_src = d_ei;
    const int* e_dst = d_ei + E;

    radix_select<<<1, 256, 0, stream>>>(d_s01, n1, d_s2, n2, k, thr);

    dim3 fillg(2048), fillb(256);
    long eb = (E + 255) / 256;
    long ab = (E + 7) / 8;
    long cvtb = ((long)N * 64 + 255) / 256;     // N*128/2 pairs
    int  gemmg = Npad / 64;

    const float* xin = d_x;
    for (int i = 0; i < 2; ++i) {
        const int HD = 128, D = 32, CT = 8;
        float* agg = (i == 0) ? xA : xB;

        pack_W<<<(4 * CT * 32 + 255) / 256, 256, 0, stream>>>(
            d_w01 + (long)i * 128 * 128, d_s01 + (long)i * 128 * 128, thr, 128, HD, CT, packw);

        cvt_bf16<<<cvtb, 256, 0, stream>>>(xin, (unsigned*)xbf, (long)N * 64);
        gemm_wmma<<<gemmg, dim3(32, CT), 0, stream>>>(xbf, packw, hbuf, N, HD, CT);

        attn_scores<<<(N * H + 255) / 256, 256, 0, stream>>>(
            hbuf, d_att01 + (long)(i * 2 + 0) * 128, d_att01 + (long)(i * 2 + 1) * 128,
            a_src, a_dst, N, H, D);

        fill_f32<<<fillg, fillb, 0, stream>>>(mbuf, -INFINITY, (long)N * H);
        fill_f32<<<fillg, fillb, 0, stream>>>(den, 0.0f, (long)N * H);
        fill_f32<<<fillg, fillb, 0, stream>>>(agg, 0.0f, (long)N * HD);

        edge_max<<<eb, 256, 0, stream>>>(e_src, e_dst, a_src, a_dst, mbuf, E, H);
        m_fix<<<(N * H + 255) / 256, 256, 0, stream>>>(mbuf, (long)N * H);
        edge_den<<<eb, 256, 0, stream>>>(e_src, e_dst, a_src, a_dst, mbuf, den, E, H);
        edge_agg<<<ab, 256, 0, stream>>>(e_src, e_dst, a_src, a_dst, mbuf, den, hbuf,
                                         agg, E, H, HD, HD / 32, 1.0f, HD, HD);

        fill_f32<<<1, 256, 0, stream>>>(bnsum, 0.0f, 128);
        fill_f32<<<1, 256, 0, stream>>>(bnsq, 0.0f, 128);
        bn_stats<<<256, 256, 0, stream>>>(agg, bnsum, bnsq, N);
        bn_apply<<<((long)N * 128 + 255) / 256, 256, 0, stream>>>(
            agg, bnsum, bnsq, d_g + (long)i * 128, d_b + (long)i * 128, N);

        xin = agg;
    }

    // final layer: HD = 160 (H=4, D=40), mean over heads folded into scale=0.25
    {
        const int HD = 160, D = 40, CT = 10;
        float* out = (float*)d_out;

        pack_W<<<(4 * CT * 32 + 255) / 256, 256, 0, stream>>>(d_w2, d_s2, thr, 128, HD, CT, packw);
        cvt_bf16<<<cvtb, 256, 0, stream>>>(xin, (unsigned*)xbf, (long)N * 64);
        gemm_wmma<<<gemmg, dim3(32, CT), 0, stream>>>(xbf, packw, hbuf, N, HD, CT);

        attn_scores<<<(N * H + 255) / 256, 256, 0, stream>>>(
            hbuf, d_att2, d_att2 + H * D, a_src, a_dst, N, H, D);

        fill_f32<<<fillg, fillb, 0, stream>>>(mbuf, -INFINITY, (long)N * H);
        fill_f32<<<fillg, fillb, 0, stream>>>(den, 0.0f, (long)N * H);
        fill_f32<<<fillg, fillb, 0, stream>>>(out, 0.0f, (long)N * D);

        edge_max<<<eb, 256, 0, stream>>>(e_src, e_dst, a_src, a_dst, mbuf, E, H);
        m_fix<<<(N * H + 255) / 256, 256, 0, stream>>>(mbuf, (long)N * H);
        edge_den<<<eb, 256, 0, stream>>>(e_src, e_dst, a_src, a_dst, mbuf, den, E, H);
        edge_agg<<<ab, 256, 0, stream>>>(e_src, e_dst, a_src, a_dst, mbuf, den, hbuf,
                                         out, E, H, HD, HD / 32, 0.25f, D, D);
    }
}
